// PrimedGKALayer_16690242912585
// MI455X (gfx1250) — matrix operand order repaired and checked by reading, not compiled, not run on hardware
//
#include <hip/hip_runtime.h>
#include <hip/hip_bf16.h>
#include <math.h>

// ---------------- config (matches reference) ----------------
#define T_LEN   1024
#define D_DIM   1024
#define HQ      16
#define HKV     4
#define HK      64
#define HV      64
#define KW      4
#define C_QKV   (HQ*HK + HKV*HK + HKV*HV)   // 1536
#define CHEB_DAMP 0.25f
#define EPS_F   1e-6f
#define L_CHK   16                           // scan chunk length
#define N_CHK   (T_LEN / L_CHK)              // 64 chunks

typedef __attribute__((ext_vector_type(16))) _Float16 f16x16;
typedef __attribute__((ext_vector_type(8)))  _Float16 f16x8;
typedef __attribute__((ext_vector_type(8)))  float    f32x8;

__device__ __forceinline__ f16x16 cat8(f16x8 lo, f16x8 hi) {
    return __builtin_shufflevector(lo, hi, 0,1,2,3,4,5,6,7,8,9,10,11,12,13,14,15);
}

// ---- fragment builders from f32 LDS (f16 convert on the fly) ----
// A 16x32 layout: lane=row, element e -> K = k0 + (e&7) + 16*(e>>3) + 8*khalf
__device__ __forceinline__ f16x16 afrag_rm(const float* b, int ld, int row, int k0, int khalf) {
    f16x16 a;
#pragma unroll
    for (int e = 0; e < 16; ++e) {
        int kk = k0 + (e & 7) + ((e >> 3) << 4) + (khalf << 3);
        a[e] = (_Float16)b[row * ld + kk];
    }
    return a;
}
// A where logical A[m][kk] = X[kk][m]  (X row-major, ld) -- used for K'^T
__device__ __forceinline__ f16x16 afrag_tr(const float* b, int ld, int row, int khalf) {
    f16x16 a;
#pragma unroll
    for (int e = 0; e < 16; ++e) {
        int kk = (e & 7) + ((e >> 3) << 4) + (khalf << 3);
        a[e] = (_Float16)b[kk * ld + row];
    }
    return a;
}
// B 32x16 layout: lane=col, element e -> K = k0 + (e&7) + 16*(e>>3) + 8*khalf.  B row-major [d][n].
__device__ __forceinline__ f16x16 bfrag_rm(const float* b, int ld, int col, int k0, int khalf) {
    f16x16 v;
#pragma unroll
    for (int e = 0; e < 16; ++e) {
        int kk = k0 + (e & 7) + ((e >> 3) << 4) + (khalf << 3);
        v[e] = (_Float16)b[kk * ld + col];
    }
    return v;
}
// B = X^T with X row-major [n][d]: contiguous per-lane reads
__device__ __forceinline__ f16x16 bfrag_tr(const float* b, int ld, int col, int k0, int khalf) {
    f16x16 v;
#pragma unroll
    for (int e = 0; e < 16; ++e) {
        int kk = k0 + (e & 7) + ((e >> 3) << 4) + (khalf << 3);
        v[e] = (_Float16)b[col * ld + kk];
    }
    return v;
}

// ---------------------------------------------------------------------------
// LDS-staged WMMA GEMM (unchanged from round 2): C = A @ W, f16 in / f32 acc.
// Block 128 threads (4 waves), tile 64x64, K step 32, 4 WMMA per wave-step.
// ---------------------------------------------------------------------------
__global__ __launch_bounds__(128)
void wmma_gemm_f16(const float* __restrict__ A, const float* __restrict__ W,
                   float* __restrict__ C, int K, int N, int ldc, int col_off)
{
    const int m0   = blockIdx.x * 64;
    const int n0   = blockIdx.y * 64;
    const int tid  = threadIdx.x;
    const int wave = tid >> 5;
    const int lane = tid & 31;
    const int rlo  = lane & 15;
    const int khalf = lane >> 4;

    __shared__ __align__(16) _Float16 As[64][40];
    __shared__ __align__(16) _Float16 Bs[64][40];

    f32x8 acc[4] = {};

    for (int k0 = 0; k0 < K; k0 += 32) {
#pragma unroll
        for (int i = 0; i < 4; ++i) {
            int idx = tid + i * 128;
            int r  = idx >> 3;
            int kq = idx & 7;
            float4 fa = *(const float4*)(A + (size_t)(m0 + r) * K + k0 + kq * 4);
            As[r][kq * 4 + 0] = (_Float16)fa.x;
            As[r][kq * 4 + 1] = (_Float16)fa.y;
            As[r][kq * 4 + 2] = (_Float16)fa.z;
            As[r][kq * 4 + 3] = (_Float16)fa.w;
            int kr = idx >> 4;
            int nq = idx & 15;
            float4 fb = *(const float4*)(W + (size_t)(k0 + kr) * N + n0 + nq * 4);
            Bs[nq * 4 + 0][kr] = (_Float16)fb.x;
            Bs[nq * 4 + 1][kr] = (_Float16)fb.y;
            Bs[nq * 4 + 2][kr] = (_Float16)fb.z;
            Bs[nq * 4 + 3][kr] = (_Float16)fb.w;
        }
        if (k0 + 32 < K) {
            __builtin_prefetch(A + (size_t)(m0 + (tid >> 1)) * K + (k0 + 32) + (tid & 1) * 16, 0, 1);
            __builtin_prefetch(W + (size_t)(k0 + 32 + (tid >> 2)) * N + n0 + (tid & 3) * 16, 0, 1);
        }
        __syncthreads();

        const int arow = wave * 16 + rlo;
        f16x16 afrag = cat8(*(const f16x8*)&As[arow][khalf * 8],
                            *(const f16x8*)&As[arow][16 + khalf * 8]);
#pragma unroll
        for (int nt = 0; nt < 4; ++nt) {
            const int bcol = nt * 16 + rlo;
            f16x16 bfrag = cat8(*(const f16x8*)&Bs[bcol][khalf * 8],
                                *(const f16x8*)&Bs[bcol][16 + khalf * 8]);
            acc[nt] = __builtin_amdgcn_wmma_f32_16x16x32_f16(
                          false, afrag, false, bfrag, (short)0, acc[nt], false, false);
        }
        __syncthreads();
    }

    const int row0 = m0 + wave * 16 + (khalf << 3);
#pragma unroll
    for (int nt = 0; nt < 4; ++nt) {
        const int col = col_off + n0 + nt * 16 + rlo;
#pragma unroll
        for (int j = 0; j < 8; ++j)
            C[(size_t)(row0 + j) * ldc + col] = acc[nt][j];
    }
}

// ---------------------------------------------------------------------------
// Gating projections + pointwise gating math (unchanged).
// ---------------------------------------------------------------------------
__global__ __launch_bounds__(256)
void gka_gating(const float* __restrict__ x,
                const float* __restrict__ Wa, const float* __restrict__ Walpha,
                const float* __restrict__ Wb, const float* __restrict__ A_log,
                const float* __restrict__ dt_bias,
                float* __restrict__ g, float* __restrict__ alpha,
                float* __restrict__ beta)
{
    int idx = blockIdx.x * blockDim.x + threadIdx.x;
    if (idx >= T_LEN * (HQ + HQ + HKV)) return;
    int t = idx / (HQ + HQ + HKV);
    int j = idx % (HQ + HQ + HKV);
    const float* xr = x + (size_t)t * D_DIM;
    float s = 0.f;
    if (j < HQ) {
        for (int kk = 0; kk < D_DIM; ++kk) s += xr[kk] * Wa[kk * HQ + j];
        float z  = s + dt_bias[j];
        float sp = (z > 20.f) ? z : log1pf(__expf(z));
        g[t * HQ + j] = -__expf(A_log[j]) * sp;
    } else if (j < 2 * HQ) {
        int c = j - HQ;
        for (int kk = 0; kk < D_DIM; ++kk) s += xr[kk] * Walpha[kk * HQ + c];
        alpha[t * HQ + c] = 1.f / (1.f + __expf(-s));
    } else {
        int c = j - 2 * HQ;
        for (int kk = 0; kk < D_DIM; ++kk) s += xr[kk] * Wb[kk * HKV + c];
        beta[t * HKV + c] = 1.f / (1.f + __expf(-s));
    }
}

// ---------------------------------------------------------------------------
// Causal conv (KW=4) + SiLU + split/normalize (unchanged).
// ---------------------------------------------------------------------------
__global__ __launch_bounds__(256)
void conv_silu_split(const float* __restrict__ qkv,
                     const float* __restrict__ Wconv,
                     const float* __restrict__ beta,
                     float* __restrict__ q, float* __restrict__ k,
                     float* __restrict__ v)
{
    const int t   = blockIdx.x;
    const int tid = threadIdx.x;
    __shared__ float buf[C_QKV];
    __shared__ float kinv[HKV];

    for (int c = tid; c < C_QKV; c += 256) {
        float acc = 0.f;
#pragma unroll
        for (int i = 0; i < KW; ++i) {
            int ts = t + i - (KW - 1);
            float xv = (ts >= 0) ? qkv[(size_t)ts * C_QKV + c] : 0.f;
            acc += xv * Wconv[c * KW + i];
        }
        buf[c] = acc / (1.f + __expf(-acc));
    }
    __syncthreads();

    if (tid < HKV) {
        float ss = 0.f;
        for (int j = 0; j < HK; ++j) {
            float kv_ = buf[HQ * HK + tid * HK + j];
            ss += kv_ * kv_;
        }
        kinv[tid] = 1.f / (sqrtf(ss) + EPS_F);
    }
    __syncthreads();

    const float qscale = 0.125f;
    for (int c = tid; c < C_QKV; c += 256) {
        if (c < HQ * HK) {
            q[(size_t)t * (HQ * HK) + c] = buf[c] * qscale;
        } else if (c < HQ * HK + HKV * HK) {
            int cc = c - HQ * HK;
            k[(size_t)t * (HKV * HK) + cc] = buf[c] * kinv[cc >> 6];
        } else {
            int cc = c - (HQ * HK + HKV * HK);
            v[(size_t)t * (HKV * HV) + cc] = buf[c] * beta[t * HKV + (cc >> 6)];
        }
    }
}

// ---------------------------------------------------------------------------
// CHUNKED gated dual-state scan, WMMA throughout.  One block (8 waves) / head.
// Chunk L=16:  c_i = within-chunk cumsum of g  (all exp factors <= 1).
//   Y  = (e^c Q) S0 + (M.(Q K^T) e^{ci-cj}) K           ( = q S per row )
//   Z  = (e^c Y) S0 + (M.(Y K^T) e^{ci-cj}) K           ( = (qS) S )
//   Xq = Q - damp*Y + damp^2*Z
//   O  = (e^c Xq) U0 + (M.(Xq K^T) e^{ci-cj}) V
//   S <- e^{cL} S + K'^T K ;  U <- e^{cL} U + K'^T V,  K' = e^{cL-cj} K
// All contractions are 16x16 WMMA tiles; L=16 dims zero-padded to 32.
// ---------------------------------------------------------------------------
__global__ __launch_bounds__(256)
void gka_scan_chunked(const float* __restrict__ qg, const float* __restrict__ kg,
                      const float* __restrict__ vg, const float* __restrict__ gg,
                      const float* __restrict__ alpha,
                      const float* __restrict__ norm_w,
                      float* __restrict__ o)
{
    const int h    = blockIdx.x;
    const int kvh  = h >> 2;
    const int tid  = threadIdx.x;
    const int wave = tid >> 5;
    const int lane = tid & 31;
    const int rlo  = lane & 15;
    const int khalf = lane >> 4;

    __shared__ float S[64 * 64];      // Hkk state
    __shared__ float U[64 * 64];      // Hkv state
    __shared__ float Kc[32 * 64];     // chunk K, rows 16..31 zero
    __shared__ float Vc[32 * 64];     // chunk V, rows 16..31 zero
    __shared__ float Kp[32 * 64];     // K' = e^{cL-cj} K, rows 16..31 zero
    __shared__ float Qb[16 * 64];     // chunk Q (raw)
    __shared__ float Xs[16 * 64];     // diag(e^c) * current operand
    __shared__ float Yb[16 * 64];     // Y, later reused as O
    __shared__ float Zb[16 * 64];
    __shared__ float Xq[16 * 64];
    __shared__ float Pm[16 * 32];     // masked/scaled P, cols 16..31 zero
    __shared__ float csum[L_CHK], aexp[L_CHK], kpsc[L_CHK], rn[L_CHK];
    __shared__ float eCL;

    for (int i = tid; i < 64 * 64; i += 256) { S[i] = 0.f; U[i] = 0.f; }
    for (int i = 16 * 64 + tid; i < 32 * 64; i += 256) { Kc[i] = 0.f; Vc[i] = 0.f; Kp[i] = 0.f; }
    __syncthreads();

    for (int ch = 0; ch < N_CHK; ++ch) {
        const int t0 = ch * L_CHK;

        // ---- load chunk q/k/v ----
        for (int idx = tid; idx < 16 * 64; idx += 256) {
            int i = idx >> 6, c = idx & 63;
            Qb[idx] = qg[((size_t)(t0 + i) * HQ + h) * HK + c];
            Kc[idx] = kg[((size_t)(t0 + i) * HKV + kvh) * HK + c];
            Vc[idx] = vg[((size_t)(t0 + i) * HKV + kvh) * HV + c];
        }
        __syncthreads();

        if (tid == 0) {
            float run = 0.f;
            for (int i = 0; i < L_CHK; ++i) {
                run += gg[(t0 + i) * HQ + h];
                csum[i] = run;
                aexp[i] = __expf(run);
            }
            for (int i = 0; i < L_CHK; ++i) kpsc[i] = __expf(csum[L_CHK - 1] - csum[i]);
            eCL = __expf(csum[L_CHK - 1]);
        }
        __syncthreads();

        // ---- K' and Xs = e^c * Q ----
        for (int idx = tid; idx < 16 * 64; idx += 256) {
            int i = idx >> 6;
            Kp[idx] = Kc[idx] * kpsc[i];
            Xs[idx] = Qb[idx] * aexp[i];
        }
        __syncthreads();

        // ========== Y pass ==========
        if (wave < 4) {            // Y0 = Xs @ S   (16x64x64)
            f32x8 acc = {};
#pragma unroll
            for (int k0 = 0; k0 < 64; k0 += 32) {
                f16x16 a = afrag_rm(Xs, 64, rlo, k0, khalf);
                f16x16 b = bfrag_rm(S, 64, wave * 16 + rlo, k0, khalf);
                acc = __builtin_amdgcn_wmma_f32_16x16x32_f16(false, a, false, b, (short)0, acc, false, false);
            }
#pragma unroll
            for (int j = 0; j < 8; ++j)
                Yb[(j + khalf * 8) * 64 + wave * 16 + rlo] = acc[j];
        } else if (wave == 4) {    // P = Q @ K^T   (16x64x16)
            f32x8 acc = {};
#pragma unroll
            for (int k0 = 0; k0 < 64; k0 += 32) {
                f16x16 a = afrag_rm(Qb, 64, rlo, k0, khalf);
                f16x16 b = bfrag_tr(Kc, 64, rlo, k0, khalf);
                acc = __builtin_amdgcn_wmma_f32_16x16x32_f16(false, a, false, b, (short)0, acc, false, false);
            }
#pragma unroll
            for (int j = 0; j < 8; ++j)
                Pm[(j + khalf * 8) * 32 + rlo] = acc[j];
        }
        __syncthreads();
        // mask + decay-scale P
        for (int idx = tid; idx < 16 * 32; idx += 256) {
            int i = idx >> 5, j = idx & 31;
            float pv = (j < 16 && j <= i) ? Pm[idx] * __expf(csum[i] - csum[j]) : 0.f;
            Pm[idx] = pv;
        }
        __syncthreads();
        if (wave < 4) {            // Y += Pm @ K   (16x32x64, rows 16..31 zero)
            f32x8 cin;
#pragma unroll
            for (int j = 0; j < 8; ++j) cin[j] = Yb[(j + khalf * 8) * 64 + wave * 16 + rlo];
            f16x16 a = afrag_rm(Pm, 32, rlo, 0, khalf);
            f16x16 b = bfrag_rm(Kc, 64, wave * 16 + rlo, 0, khalf);
            f32x8 acc = __builtin_amdgcn_wmma_f32_16x16x32_f16(false, a, false, b, (short)0, cin, false, false);
#pragma unroll
            for (int j = 0; j < 8; ++j) Yb[(j + khalf * 8) * 64 + wave * 16 + rlo] = acc[j];
        }
        __syncthreads();

        // ========== Z pass ==========
        for (int idx = tid; idx < 16 * 64; idx += 256) Xs[idx] = Yb[idx] * aexp[idx >> 6];
        __syncthreads();
        if (wave < 4) {            // Z0 = Xs @ S
            f32x8 acc = {};
#pragma unroll
            for (int k0 = 0; k0 < 64; k0 += 32) {
                f16x16 a = afrag_rm(Xs, 64, rlo, k0, khalf);
                f16x16 b = bfrag_rm(S, 64, wave * 16 + rlo, k0, khalf);
                acc = __builtin_amdgcn_wmma_f32_16x16x32_f16(false, a, false, b, (short)0, acc, false, false);
            }
#pragma unroll
            for (int j = 0; j < 8; ++j) Zb[(j + khalf * 8) * 64 + wave * 16 + rlo] = acc[j];
        } else if (wave == 4) {    // P = Y @ K^T
            f32x8 acc = {};
#pragma unroll
            for (int k0 = 0; k0 < 64; k0 += 32) {
                f16x16 a = afrag_rm(Yb, 64, rlo, k0, khalf);
                f16x16 b = bfrag_tr(Kc, 64, rlo, k0, khalf);
                acc = __builtin_amdgcn_wmma_f32_16x16x32_f16(false, a, false, b, (short)0, acc, false, false);
            }
#pragma unroll
            for (int j = 0; j < 8; ++j) Pm[(j + khalf * 8) * 32 + rlo] = acc[j];
        }
        __syncthreads();
        for (int idx = tid; idx < 16 * 32; idx += 256) {
            int i = idx >> 5, j = idx & 31;
            float pv = (j < 16 && j <= i) ? Pm[idx] * __expf(csum[i] - csum[j]) : 0.f;
            Pm[idx] = pv;
        }
        __syncthreads();
        if (wave < 4) {            // Z += Pm @ K
            f32x8 cin;
#pragma unroll
            for (int j = 0; j < 8; ++j) cin[j] = Zb[(j + khalf * 8) * 64 + wave * 16 + rlo];
            f16x16 a = afrag_rm(Pm, 32, rlo, 0, khalf);
            f16x16 b = bfrag_rm(Kc, 64, wave * 16 + rlo, 0, khalf);
            f32x8 acc = __builtin_amdgcn_wmma_f32_16x16x32_f16(false, a, false, b, (short)0, cin, false, false);
#pragma unroll
            for (int j = 0; j < 8; ++j) Zb[(j + khalf * 8) * 64 + wave * 16 + rlo] = acc[j];
        }
        __syncthreads();

        // ========== Xq + O pass ==========
        for (int idx = tid; idx < 16 * 64; idx += 256) {
            float xqv = Qb[idx] - CHEB_DAMP * Yb[idx] + (CHEB_DAMP * CHEB_DAMP) * Zb[idx];
            Xq[idx] = xqv;
            Xs[idx] = xqv * aexp[idx >> 6];
        }
        __syncthreads();
        if (wave < 4) {            // O0 = Xs @ U   (into Yb, Y is dead)
            f32x8 acc = {};
#pragma unroll
            for (int k0 = 0; k0 < 64; k0 += 32) {
                f16x16 a = afrag_rm(Xs, 64, rlo, k0, khalf);
                f16x16 b = bfrag_rm(U, 64, wave * 16 + rlo, k0, khalf);
                acc = __builtin_amdgcn_wmma_f32_16x16x32_f16(false, a, false, b, (short)0, acc, false, false);
            }
#pragma unroll
            for (int j = 0; j < 8; ++j) Yb[(j + khalf * 8) * 64 + wave * 16 + rlo] = acc[j];
        } else if (wave == 4) {    // P = Xq @ K^T
            f32x8 acc = {};
#pragma unroll
            for (int k0 = 0; k0 < 64; k0 += 32) {
                f16x16 a = afrag_rm(Xq, 64, rlo, k0, khalf);
                f16x16 b = bfrag_tr(Kc, 64, rlo, k0, khalf);
                acc = __builtin_amdgcn_wmma_f32_16x16x32_f16(false, a, false, b, (short)0, acc, false, false);
            }
#pragma unroll
            for (int j = 0; j < 8; ++j) Pm[(j + khalf * 8) * 32 + rlo] = acc[j];
        }
        __syncthreads();
        for (int idx = tid; idx < 16 * 32; idx += 256) {
            int i = idx >> 5, j = idx & 31;
            float pv = (j < 16 && j <= i) ? Pm[idx] * __expf(csum[i] - csum[j]) : 0.f;
            Pm[idx] = pv;
        }
        __syncthreads();
        if (wave < 4) {            // O += Pm @ V
            f32x8 cin;
#pragma unroll
            for (int j = 0; j < 8; ++j) cin[j] = Yb[(j + khalf * 8) * 64 + wave * 16 + rlo];
            f16x16 a = afrag_rm(Pm, 32, rlo, 0, khalf);
            f16x16 b = bfrag_rm(Vc, 64, wave * 16 + rlo, 0, khalf);
            f32x8 acc = __builtin_amdgcn_wmma_f32_16x16x32_f16(false, a, false, b, (short)0, cin, false, false);
#pragma unroll
            for (int j = 0; j < 8; ++j) Yb[(j + khalf * 8) * 64 + wave * 16 + rlo] = acc[j];
        }
        __syncthreads();

        // ---- alpha gate + RMSNorm + store ----
        for (int idx = tid; idx < 16 * 64; idx += 256) {
            int i = idx >> 6;
            Yb[idx] *= alpha[(t0 + i) * HQ + h];
        }
        __syncthreads();
        if (tid < L_CHK) {
            float ss = 0.f;
            for (int c = 0; c < 64; ++c) { float ov = Yb[tid * 64 + c]; ss += ov * ov; }
            rn[tid] = rsqrtf(ss / 64.f + EPS_F);
        }
        __syncthreads();
        for (int idx = tid; idx < 16 * 64; idx += 256) {
            int i = idx >> 6, c = idx & 63;
            o[((size_t)(t0 + i) * HQ + h) * HV + c] = Yb[idx] * rn[i] * norm_w[c];
        }

        // ---- state update: S <- eCL*S + K'^T K ; U <- eCL*U + K'^T V ----
        // 16 tiles each, 2 waves-worth per wave (all 8 waves), K=16 padded to 32.
        const float ecl = eCL;
#pragma unroll
        for (int rep = 0; rep < 2; ++rep) {
            int ti = wave * 2 + rep;        // 0..15
            int mi = ti >> 2, ni = ti & 3;
            f16x16 a = afrag_tr(Kp, 64, mi * 16 + rlo, khalf);
            {   // S tile
                f16x16 b = bfrag_rm(Kc, 64, ni * 16 + rlo, 0, khalf);
                f32x8 cin;
#pragma unroll
                for (int j = 0; j < 8; ++j)
                    cin[j] = ecl * S[(mi * 16 + j + khalf * 8) * 64 + ni * 16 + rlo];
                f32x8 acc = __builtin_amdgcn_wmma_f32_16x16x32_f16(false, a, false, b, (short)0, cin, false, false);
#pragma unroll
                for (int j = 0; j < 8; ++j)
                    S[(mi * 16 + j + khalf * 8) * 64 + ni * 16 + rlo] = acc[j];
            }
            {   // U tile
                f16x16 b = bfrag_rm(Vc, 64, ni * 16 + rlo, 0, khalf);
                f32x8 cin;
#pragma unroll
                for (int j = 0; j < 8; ++j)
                    cin[j] = ecl * U[(mi * 16 + j + khalf * 8) * 64 + ni * 16 + rlo];
                f32x8 acc = __builtin_amdgcn_wmma_f32_16x16x32_f16(false, a, false, b, (short)0, cin, false, false);
#pragma unroll
                for (int j = 0; j < 8; ++j)
                    U[(mi * 16 + j + khalf * 8) * 64 + ni * 16 + rlo] = acc[j];
            }
        }
        __syncthreads();   // next chunk overwrites Kc/Vc/Qb
    }
}

// ---------------------------------------------------------------------------
extern "C" void kernel_launch(void* const* d_in, const int* in_sizes, int n_in,
                              void* d_out, int out_size, void* d_ws, size_t ws_size,
                              hipStream_t stream)
{
    const float* x       = (const float*)d_in[0];
    const float* Wq      = (const float*)d_in[1];
    const float* Wk      = (const float*)d_in[2];
    const float* Wv      = (const float*)d_in[3];
    const float* Wconv   = (const float*)d_in[4];
    const float* Wa      = (const float*)d_in[5];
    const float* Walpha  = (const float*)d_in[6];
    const float* Wb      = (const float*)d_in[7];
    const float* A_log   = (const float*)d_in[8];
    const float* dt_bias = (const float*)d_in[9];
    const float* norm_w  = (const float*)d_in[10];
    const float* Wo      = (const float*)d_in[11];
    float*       out     = (float*)d_out;

    float* ws     = (float*)d_ws;
    size_t off    = 0;
    float* qkv    = ws + off; off += (size_t)T_LEN * C_QKV;
    float* gbuf   = ws + off; off += (size_t)T_LEN * HQ;
    float* abuf   = ws + off; off += (size_t)T_LEN * HQ;
    float* bbuf   = ws + off; off += (size_t)T_LEN * HKV;
    float* qb     = ws + off; off += (size_t)T_LEN * HQ * HK;
    float* kb     = ws + off; off += (size_t)T_LEN * HKV * HK;
    float* vb     = ws + off; off += (size_t)T_LEN * HKV * HV;
    float* ob     = ws + off; off += (size_t)T_LEN * HQ * HV;

    wmma_gemm_f16<<<dim3(T_LEN/64, (HQ*HK)/64), 128, 0, stream>>>(
        x, Wq, qkv, D_DIM, HQ*HK, C_QKV, 0);
    wmma_gemm_f16<<<dim3(T_LEN/64, (HKV*HK)/64), 128, 0, stream>>>(
        x, Wk, qkv, D_DIM, HKV*HK, C_QKV, HQ*HK);
    wmma_gemm_f16<<<dim3(T_LEN/64, (HKV*HV)/64), 128, 0, stream>>>(
        x, Wv, qkv, D_DIM, HKV*HV, C_QKV, HQ*HK + HKV*HK);

    {
        int total = T_LEN * (HQ + HQ + HKV);
        gka_gating<<<(total + 255) / 256, 256, 0, stream>>>(
            x, Wa, Walpha, Wb, A_log, dt_bias, gbuf, abuf, bbuf);
    }

    conv_silu_split<<<T_LEN, 256, 0, stream>>>(qkv, Wconv, bbuf, qb, kb, vb);

    gka_scan_chunked<<<HQ, 256, 0, stream>>>(qb, kb, vb, gbuf, abuf, norm_w, ob);

    wmma_gemm_f16<<<dim3(T_LEN/64, D_DIM/64), 128, 0, stream>>>(
        ob, Wo, out, HQ*HV, D_DIM, D_DIM, 0);
}
